// SLConv_10307921510659
// MI455X (gfx1250) — matrix operand order repaired and checked by reading, not compiled, not run on hardware
//
#include <hip/hip_runtime.h>

typedef __attribute__((ext_vector_type(16))) _Float16 v16h;
typedef __attribute__((ext_vector_type(4)))  _Float16 v4h;
typedef __attribute__((ext_vector_type(8)))  float    v8f;
typedef __attribute__((ext_vector_type(4)))  uint32_t v4u;

#define NH     256      // h
#define NL     4096     // L
#define NB     16       // batch
#define KLEN   1024     // built kernel length
#define LCHUNK 256      // l positions per workgroup
#define NWAVES 4
#define NTILES 4        // 16-wide l-tiles per wave  (NWAVES*NTILES*16 == LCHUNK)
#define XWIN   1304     // x window halves (needs >= 240 + 1056); 1304 -> conflict-free b128
#define KPAD   16
#define KRSZ   1072     // KPAD + 1024 + tail pad; 536 dwords per channel
#define KSTEPS 33       // 33*32 = 1056 >= 1040 band width

#define XS_OFF   0
#define KR0_OFF  (NB * XWIN * 2)                  /* 41728 */
#define KR1_OFF  (KR0_OFF + 4 * KRSZ * 2)         /* +8576 */
#define NRM_OFF  (KR1_OFF + 4 * KRSZ * 2)         /* +8576 */
#define LDS_BYTES (NRM_OFF + 16)                  /* 58896 < 64KB */

static_assert(NWAVES * NTILES * 16 == LCHUNK, "tiling mismatch");
static_assert((XWIN & 3) == 0, "XWIN must be multiple of 4");

// decode built-kernel index -> (scale i, tap j, multiplier 2^(5-i))
__device__ __forceinline__ void kdecode(int idx, int& i, int& j, float& scale) {
  if (idx < 64) { i = idx >> 5; j = idx & 31; }
  else {
    int lz = 31 - __clz(idx);          // 6..9
    i = lz - 4;                        // 2..5
    j = (idx - (1 << lz)) >> (lz - 5); // repeat factor 2^(i-1)
  }
  scale = (float)(1 << (5 - i));
}

__global__ __launch_bounds__(128)
void slconv_wmma(const float* __restrict__ x, const float* __restrict__ kern,
                 const float* __restrict__ Dm, float* __restrict__ out) {
  extern __shared__ char smem[];
  _Float16* xs   = (_Float16*)(smem + XS_OFF);   // [16][XWIN] f16 window of x
  _Float16* kr0  = (_Float16*)(smem + KR0_OFF);  // [4][KRSZ] reversed+padded norm'd k
  _Float16* kr1  = (_Float16*)(smem + KR1_OFF);  // kr1[i] = kr0[i+1] (odd-parity copy)
  float*    nacc = (float*)   (smem + NRM_OFF);  // [4] norm accumulators

  const int tid = threadIdx.x;          // 128 threads = 4 waves
  const int h   = blockIdx.x;           // 0..255
  const int l0  = blockIdx.y * LCHUNK;  // chunk start in l

  if (tid < 4) nacc[tid] = 0.f;

  // ---- stage x window into LDS as f16: xs[b][u] = x_hat[b,h, l0-512+u] ----
  // l0 multiple of 256, u multiple of 4 => p multiple of 4 => float4 loads are 16B-aligned
  for (int b = 0; b < NB; ++b) {
    const float* xrow = x + ((size_t)b * NH + h) * NL;
    _Float16* xsrow = xs + b * XWIN;
    for (int u = tid * 4; u < XWIN; u += 512) {
      const int p = l0 - 512 + u;
      float v0, v1, v2, v3;
      if (p >= 0 && p <= NL - 4) {
        const float4 f = *(const float4*)(xrow + p);
        v0 = f.x; v1 = f.y; v2 = f.z; v3 = f.w;
      } else {
        v0 = ((unsigned)(p + 0) < (unsigned)NL) ? xrow[p + 0] : 0.f;
        v1 = ((unsigned)(p + 1) < (unsigned)NL) ? xrow[p + 1] : 0.f;
        v2 = ((unsigned)(p + 2) < (unsigned)NL) ? xrow[p + 2] : 0.f;
        v3 = ((unsigned)(p + 3) < (unsigned)NL) ? xrow[p + 3] : 0.f;
      }
      v4h hv = { (_Float16)v0, (_Float16)v1, (_Float16)v2, (_Float16)v3 };
      *(v4h*)(xsrow + u) = hv;   // 8B-aligned ds_store_b64
    }
  }
  __syncthreads();

  // ---- pass 1: per-(c) L2 norm of built kernel (recompute, no f32 staging) ----
  float p0 = 0.f, p1 = 0.f, p2 = 0.f, p3 = 0.f;
  for (int idx = tid; idx < KLEN; idx += 128) {
    int i, j; float sc; kdecode(idx, i, j, sc);
    const float* kb = kern + (size_t)i * 32768 + (size_t)h * 32 + j; // (i,c,h,j), c-stride 8192
    float v0 = kb[0] * sc, v1 = kb[8192] * sc, v2 = kb[16384] * sc, v3 = kb[24576] * sc;
    p0 += v0 * v0; p1 += v1 * v1; p2 += v2 * v2; p3 += v3 * v3;
  }
  atomicAdd(&nacc[0], p0); atomicAdd(&nacc[1], p1);
  atomicAdd(&nacc[2], p2); atomicAdd(&nacc[3], p3);
  __syncthreads();
  float rn[4] = { rsqrtf(nacc[0]), rsqrtf(nacc[1]), rsqrtf(nacc[2]), rsqrtf(nacc[3]) };

  // ---- pass 2: reversed + padded + normalized f16 kernel, two parity copies ----
  if (tid < 4) kr1[tid * KRSZ + (KRSZ - 1)] = (_Float16)0.f;
  for (int u = tid; u < KRSZ; u += 128) {
    int src = 1039 - u;  // kr0[u] = k_hat[1039 - u]  (KPAD front zeros, tail zeros)
    float vs0 = 0.f, vs1 = 0.f, vs2 = 0.f, vs3 = 0.f;
    if ((unsigned)src < (unsigned)KLEN) {
      int i, j; float sc; kdecode(src, i, j, sc);
      const float* kb = kern + (size_t)i * 32768 + (size_t)h * 32 + j;
      vs0 = kb[0]     * sc * rn[0]; vs1 = kb[8192]  * sc * rn[1];
      vs2 = kb[16384] * sc * rn[2]; vs3 = kb[24576] * sc * rn[3];
    }
    kr0[0 * KRSZ + u] = (_Float16)vs0; kr0[1 * KRSZ + u] = (_Float16)vs1;
    kr0[2 * KRSZ + u] = (_Float16)vs2; kr0[3 * KRSZ + u] = (_Float16)vs3;
    if (u >= 1) {
      kr1[0 * KRSZ + u - 1] = (_Float16)vs0; kr1[1 * KRSZ + u - 1] = (_Float16)vs1;
      kr1[2 * KRSZ + u - 1] = (_Float16)vs2; kr1[3 * KRSZ + u - 1] = (_Float16)vs3;
    }
  }
  __syncthreads();

  // ---- main loop: D[b, l_off] += A(x, 16x32) x B(k-Toeplitz, 32x16), 4 c-accumulators ----
  const int ln   = tid & 31;
  const int wv   = tid >> 5;
  const int ncol = ln & 15;        // N = l_off (B/C/D column); also A row m=b
  const int hi   = ln >> 4;        // half-wave
  const int kupB = hi * 16;        // B: upper lanes hold K=16..31
  const int kofA = hi * 16;        // A: upper lanes hold K=16..31

  v8f acc[NTILES][4];
  {
    const v8f vz = {0.f,0.f,0.f,0.f,0.f,0.f,0.f,0.f};
    #pragma unroll
    for (int j = 0; j < NTILES; ++j)
      #pragma unroll
      for (int c = 0; c < 4; ++c) acc[j][c] = vz;
  }

  // B base: k_hat[n+1024-s] = kr0[base0 + 32t + K], base0 = KPAD-1-ncol+kupB.
  // 32t is even => parity of the lane's address is loop-invariant: hoist the
  // kr0/kr1 select; per step the dword base advances by exactly 16 dwords.
  const int base0 = KPAD - 1 - ncol + kupB;
  const _Float16* bsrc = (base0 & 1) ? kr1 : kr0;
  const uint32_t* kpt = (const uint32_t*)bsrc + (base0 >> 1);   // +16 dwords / step

  // A base: running pointer, +32 halves / step; per-tile offsets are immediates.
  const _Float16* xat = xs + ncol * XWIN + kofA;

  #pragma unroll 1
  for (int t = 0; t < KSTEPS; ++t) {
    v16h Bm[4];
    #pragma unroll
    for (int c = 0; c < 4; ++c) {                 // 8 ds_load_b32, imm offsets c*536+q
      union { uint32_t u[8]; v16h hh; } bb;
      #pragma unroll
      for (int q = 0; q < 8; ++q) bb.u[q] = kpt[c * (KRSZ / 2) + q];
      Bm[c] = bb.hh;
    }
    #pragma unroll
    for (int j = 0; j < NTILES; ++j) {
      const int tb = (wv * NTILES + j) * 16;      // imm offset per tile
      union { v4u u[2]; v16h hh; } aa;
      aa.u[0] = *(const v4u*)(xat + tb);          // K = kofA .. kofA+7
      aa.u[1] = *(const v4u*)(xat + tb + 8);      // K = kofA+8 .. kofA+15
      #pragma unroll
      for (int c = 0; c < 4; ++c)
        acc[j][c] = __builtin_amdgcn_wmma_f32_16x16x32_f16(
            false, aa.hh, false, Bm[c], (short)0, acc[j][c], false, false);
    }
    kpt += 16;       // next K-step: s32 += 32 halves = 16 dwords
    xat += 32;       // next K-step: +32 halves
  }

  // ---- epilogue: fuse skip x*D[c,h], write out[b, 4h+c, l] ----
  float Dv[4];
  #pragma unroll
  for (int c = 0; c < 4; ++c) Dv[c] = Dm[c * NH + h];
  const int rbase = hi * 8;  // D-matrix: VGPR v holds M = rbase + v
  #pragma unroll
  for (int j = 0; j < NTILES; ++j) {
    const int tb = (wv * NTILES + j) * 16;
    float xv[8];
    #pragma unroll
    for (int v = 0; v < 8; ++v)
      xv[v] = (float)xs[(rbase + v) * XWIN + tb + 512 + ncol];  // x[b,h,l]
    const int lpos = l0 + tb + ncol;
    #pragma unroll
    for (int c = 0; c < 4; ++c) {
      #pragma unroll
      for (int v = 0; v < 8; ++v) {
        const int brow = rbase + v;
        const size_t o = (((size_t)brow * (NH * 4) + (size_t)h * 4 + c) * (size_t)NL) + lpos;
        out[o] = acc[j][c][v] + xv[v] * Dv[c];
      }
    }
  }
}

extern "C" void kernel_launch(void* const* d_in, const int* in_sizes, int n_in,
                              void* d_out, int out_size, void* d_ws, size_t ws_size,
                              hipStream_t stream) {
  (void)in_sizes; (void)n_in; (void)d_ws; (void)ws_size; (void)out_size;
  const float* x    = (const float*)d_in[0];  // (16,256,4096)
  const float* kern = (const float*)d_in[1];  // (6,4,256,32)
  const float* Dm   = (const float*)d_in[2];  // (4,256)
  float* out = (float*)d_out;                 // (16,1024,4096)
  dim3 grid(NH, NL / LCHUNK, 1);              // 256 x 16 workgroups
  dim3 block(NWAVES * 32, 1, 1);              // 4 waves
  slconv_wmma<<<grid, block, LDS_BYTES, stream>>>(x, kern, Dm, out);
}
